// PolicyNetRegression_15350213116267
// MI455X (gfx1250) — matrix operand order repaired and checked by reading, not compiled, not run on hardware
//
#include <hip/hip_runtime.h>
#include <hip/hip_bf16.h>
#include <math.h>

typedef __attribute__((ext_vector_type(16))) _Float16 v16h;
typedef __attribute__((ext_vector_type(8)))  float    v8f;

#define EPSV 1e-10f

// ---------------------------------------------------------------------------
// WMMA fragment helpers (16x16x32 f16; A/B use this code's contiguous k order, same for both)
// A (16x32): lane m=lane&15, halves 0..15 -> K=hi*16+t
// B (32x16): lane n=lane&15, halves 0..15 -> K=hi*16+t  (col n of B = row n of W)
// C/D (16x16 f32): row = vgpr + hi*8, col = lane&15
// ---------------------------------------------------------------------------
__device__ inline v16h frag_a_f16(const _Float16* __restrict__ A, int lda, int row0, int k0) {
  int lane = threadIdx.x & 31;
  int lo = lane & 15, hi = lane >> 4;
  const _Float16* p = A + (size_t)(row0 + lo) * lda + k0 + hi * 16;
  v16h a;
#pragma unroll
  for (int t = 0; t < 16; ++t) a[t] = p[t];
  return a;
}

__device__ inline v16h frag_b_f16(const _Float16* __restrict__ W, int ldw, int col0, int k0) {
  int lane = threadIdx.x & 31;
  int lo = lane & 15, hi = lane >> 4;
  const _Float16* p = W + (size_t)(col0 + lo) * ldw + k0 + hi * 16;
  v16h b;
#pragma unroll
  for (int t = 0; t < 16; ++t) b[t] = p[t];
  return b;
}

__device__ inline float sigm(float x) { return 1.f / (1.f + expf(-x)); }

#define WMMA_F16(a, b, c) \
  __builtin_amdgcn_wmma_f32_16x16x32_f16(false, (a), false, (b), (short)0, (c), false, false)

// ---------------------------------------------------------------------------
// Utility kernels
// ---------------------------------------------------------------------------
__global__ __launch_bounds__(256) void cvt_f32_to_f16(const float* __restrict__ src,
                                                      _Float16* __restrict__ dst, int n) {
  int i = blockIdx.x * 256 + threadIdx.x;
  if (i < n) dst[i] = (_Float16)src[i];
}

__global__ __launch_bounds__(256) void zero_f32(float* __restrict__ p, int n) {
  int i = blockIdx.x * 256 + threadIdx.x;
  if (i < n) p[i] = 0.f;
}

// zero h (f16) and c (f32) recurrent state
__global__ __launch_bounds__(256) void init_state(_Float16* __restrict__ h16,
                                                  float* __restrict__ c, int n) {
  int i = blockIdx.x * 256 + threadIdx.x;
  if (i < n) { h16[i] = (_Float16)0.f; c[i] = 0.f; }
}

// x16[b,s,:] = (f16) graph_table[inputs[b,s] * mask[b,s], :]
__global__ __launch_bounds__(256) void embed_kernel(const int* __restrict__ inputs,
                                                    const int* __restrict__ mask,
                                                    const float* __restrict__ table,
                                                    _Float16* __restrict__ x16, int E, int n) {
  int i = blockIdx.x * 256 + threadIdx.x;
  if (i >= n) return;
  int row = i / E, e = i - row * E;
  int idx = inputs[row] * mask[row];
  x16[i] = (_Float16)table[(size_t)idx * E + e];
}

// ---------------------------------------------------------------------------
// Core pipelined 16xK x Kx64 tile: 1 row-tile x 4 col-tiles, double-buffered
// fragments so next-k loads overlap the current 4 WMMAs.
// ---------------------------------------------------------------------------
__device__ inline void wmma_tile_1x4(const _Float16* __restrict__ A, int lda, int row0,
                                     const _Float16* __restrict__ W, int ldw, int col0, int K,
                                     v8f& acc0, v8f& acc1, v8f& acc2, v8f& acc3) {
  v16h a  = frag_a_f16(A, lda, row0, 0);
  v16h b0 = frag_b_f16(W, ldw, col0,      0);
  v16h b1 = frag_b_f16(W, ldw, col0 + 16, 0);
  v16h b2 = frag_b_f16(W, ldw, col0 + 32, 0);
  v16h b3 = frag_b_f16(W, ldw, col0 + 48, 0);
  for (int k0 = 32; k0 < K; k0 += 32) {
    v16h an  = frag_a_f16(A, lda, row0, k0);
    v16h nb0 = frag_b_f16(W, ldw, col0,      k0);
    v16h nb1 = frag_b_f16(W, ldw, col0 + 16, k0);
    v16h nb2 = frag_b_f16(W, ldw, col0 + 32, k0);
    v16h nb3 = frag_b_f16(W, ldw, col0 + 48, k0);
    acc0 = WMMA_F16(a, b0, acc0);
    acc1 = WMMA_F16(a, b1, acc1);
    acc2 = WMMA_F16(a, b2, acc2);
    acc3 = WMMA_F16(a, b3, acc3);
    a = an; b0 = nb0; b1 = nb1; b2 = nb2; b3 = nb3;
  }
  acc0 = WMMA_F16(a, b0, acc0);
  acc1 = WMMA_F16(a, b1, acc1);
  acc2 = WMMA_F16(a, b2, acc2);
  acc3 = WMMA_F16(a, b3, acc3);
}

// ---------------------------------------------------------------------------
// Gate-precompute GEMM: C[rows, 2048] = A[rows,K](f16) * W^T(f16) + bias1+bias2
// ---------------------------------------------------------------------------
__global__ __launch_bounds__(32) void gemm_gates_wmma(const _Float16* __restrict__ A, int K,
                                                      const _Float16* __restrict__ W,
                                                      const float* __restrict__ bias1,
                                                      const float* __restrict__ bias2,
                                                      float* __restrict__ C, int ldc) {
  int col0 = blockIdx.x * 64;
  int row0 = blockIdx.y * 16;
  v8f acc0 = {}, acc1 = {}, acc2 = {}, acc3 = {};
  wmma_tile_1x4(A, K, row0, W, K, col0, K, acc0, acc1, acc2, acc3);
  int lane = threadIdx.x & 31;
  int lo = lane & 15, hi = lane >> 4;
  v8f accs[4] = {acc0, acc1, acc2, acc3};
#pragma unroll
  for (int n = 0; n < 4; ++n) {
    int col = col0 + n * 16 + lo;
    float bb = bias1[col] + bias2[col];
#pragma unroll
    for (int v = 0; v < 8; ++v) {
      int row = row0 + hi * 8 + v;
      C[(size_t)row * ldc + col] = accs[n][v] + bb;
    }
  }
}

// ---------------------------------------------------------------------------
// Recurrent step: Z[32,2048] = xg_t + Hprev[32,H](f16) * Whh^T(f16)
// ---------------------------------------------------------------------------
__global__ __launch_bounds__(32) void lstm_z_wmma(const _Float16* __restrict__ Hprev,
                                                  const _Float16* __restrict__ Whh, int H,
                                                  const float* __restrict__ xg_t, int xg_rowstride,
                                                  float* __restrict__ Z) {
  int col0 = blockIdx.x * 64;
  int row0 = blockIdx.y * 16;
  v8f acc0 = {}, acc1 = {}, acc2 = {}, acc3 = {};
  wmma_tile_1x4(Hprev, H, row0, Whh, H, col0, H, acc0, acc1, acc2, acc3);
  int lane = threadIdx.x & 31;
  int lo = lane & 15, hi = lane >> 4;
  v8f accs[4] = {acc0, acc1, acc2, acc3};
#pragma unroll
  for (int n = 0; n < 4; ++n) {
    int col = col0 + n * 16 + lo;
#pragma unroll
    for (int v = 0; v < 8; ++v) {
      int row = row0 + hi * 8 + v;
      Z[(size_t)row * 2048 + col] = accs[n][v] + xg_t[(size_t)row * xg_rowstride + col];
    }
  }
}

// gates i,f,g,o (torch order); h kept in f16 (WMMA operand), c in f32.
// Optionally records h into an f16 sequence (layer 0) and/or f32 sequence (layer 1).
__global__ __launch_bounds__(256) void lstm_gates(const float* __restrict__ Z,
                                                  _Float16* __restrict__ h16,
                                                  float* __restrict__ c,
                                                  _Float16* __restrict__ hseq16,
                                                  float* __restrict__ hseq32,
                                                  int H, int S, int t, int n) {
  int i = blockIdx.x * 256 + threadIdx.x;
  if (i >= n) return;
  int b = i / H, j = i - b * H;
  const float* z = Z + (size_t)b * 4 * H;
  float ig = sigm(z[j]);
  float fg = sigm(z[H + j]);
  float gg = tanhf(z[2 * H + j]);
  float og = sigm(z[3 * H + j]);
  float cn = fg * c[i] + ig * gg;
  float hn = og * tanhf(cn);
  c[i] = cn;
  h16[i] = (_Float16)hn;
  size_t so = ((size_t)b * S + t) * H + j;
  if (hseq16) hseq16[so] = (_Float16)hn;
  if (hseq32) hseq32[so] = hn;
}

// j_last[b] = sum(mask[b,:]) - 1
__global__ __launch_bounds__(32) void jlast_kernel(const int* __restrict__ mask, int S,
                                                   int* __restrict__ jlast) {
  int b = threadIdx.x;
  int s = 0;
  for (int t = 0; t < S; ++t) s += mask[b * S + t];
  jlast[b] = s - 1;
}

// out_last[b,e] = tanh(dot(h1seq[b, jlast[b], :], Wl[e,:]) + bl[e]); f32 + f16 copies
__global__ __launch_bounds__(256) void proj_kernel(const float* __restrict__ hseq,
                                                   const int* __restrict__ jlast,
                                                   const float* __restrict__ Wl,
                                                   const float* __restrict__ bl,
                                                   float* __restrict__ out_last,
                                                   _Float16* __restrict__ out_last16,
                                                   int H, int E, int S, int n) {
  int i = blockIdx.x * 256 + threadIdx.x;
  if (i >= n) return;
  int b = i / E, e = i - b * E;
  const float* hp = hseq + ((size_t)b * S + jlast[b]) * H;
  const float* w = Wl + (size_t)e * H;
  float s = bl[e];
  for (int k = 0; k < H; ++k) s += hp[k] * w[k];
  float o = tanhf(s);
  out_last[i] = o;
  out_last16[i] = (_Float16)o;
}

// per-row L2 norm (f32 input); one 256-thread block per row
__global__ __launch_bounds__(256) void rownorm_kernel(const float* __restrict__ X, int N,
                                                      float* __restrict__ out) {
  __shared__ float red[256];
  int r = blockIdx.x, tid = threadIdx.x;
  float s = 0.f;
  for (int k = tid; k < N; k += 256) {
    float v = X[(size_t)r * N + k];
    s += v * v;
  }
  red[tid] = s;
  __syncthreads();
  for (int st = 128; st > 0; st >>= 1) {
    if (tid < st) red[tid] += red[tid + st];
    __syncthreads();
  }
  if (tid == 0) out[r] = sqrtf(red[0]);
}

// fused: vnorm[r] = ||X[r,:]|| and X16[r,:] = (f16)X[r,:]  (one pass over 51 MB)
__global__ __launch_bounds__(256) void rownorm_cvt_kernel(const float* __restrict__ X, int N,
                                                          float* __restrict__ out,
                                                          _Float16* __restrict__ X16) {
  __shared__ float red[256];
  int r = blockIdx.x, tid = threadIdx.x;
  float s = 0.f;
  for (int k = tid; k < N; k += 256) {
    float v = X[(size_t)r * N + k];
    X16[(size_t)r * N + k] = (_Float16)v;
    s += v * v;
  }
  red[tid] = s;
  __syncthreads();
  for (int st = 128; st > 0; st >>= 1) {
    if (tid < st) red[tid] += red[tid + st];
    __syncthreads();
  }
  if (tid == 0) out[r] = sqrtf(red[0]);
}

// ---------------------------------------------------------------------------
// Cos GEMM: cos[b,v] = (out_last . ve[v]) / (onorm[b]*vnorm[v] + EPS)
// One wave per 16-wide column tile, both 16-row tiles of M=32 (B frag reused);
// pipelined so the streaming ve loads overlap the WMMAs.
// ---------------------------------------------------------------------------
__global__ __launch_bounds__(32) void cos_wmma(const _Float16* __restrict__ Out,
                                               const _Float16* __restrict__ VE,
                                               const float* __restrict__ onorm,
                                               const float* __restrict__ vnorm,
                                               float* __restrict__ Cos, int V, int E) {
  int col0 = blockIdx.x * 16;
  v8f acc0 = {}, acc1 = {};
  v16h b  = frag_b_f16(VE, E, col0, 0);
  v16h a0 = frag_a_f16(Out, E, 0, 0);
  v16h a1 = frag_a_f16(Out, E, 16, 0);
  for (int k0 = 32; k0 < E; k0 += 32) {
    v16h nb  = frag_b_f16(VE, E, col0, k0);
    v16h na0 = frag_a_f16(Out, E, 0, k0);
    v16h na1 = frag_a_f16(Out, E, 16, k0);
    acc0 = WMMA_F16(a0, b, acc0);
    acc1 = WMMA_F16(a1, b, acc1);
    b = nb; a0 = na0; a1 = na1;
  }
  acc0 = WMMA_F16(a0, b, acc0);
  acc1 = WMMA_F16(a1, b, acc1);
  int lane = threadIdx.x & 31;
  int lo = lane & 15, hi = lane >> 4;
  int v = col0 + lo;
  float vn = vnorm[v];
#pragma unroll
  for (int r = 0; r < 8; ++r) {
    int b0 = hi * 8 + r;
    Cos[(size_t)b0 * V + v] = acc0[r] / (onorm[b0] * vn + EPSV);
    int b1 = 16 + hi * 8 + r;
    Cos[(size_t)b1 * V + v] = acc1[r] / (onorm[b1] * vn + EPSV);
  }
}

// ---------------------------------------------------------------------------
// Losses + stable top-k membership per batch row.
// Label l is in top_k(probs,K) iff its stable rank over cos < K
// (softmax monotonic; jax top_k ties broken by lower index).
// ---------------------------------------------------------------------------
#define LBL 20
__global__ __launch_bounds__(256) void loss_kernel(const float* __restrict__ Cos,
                                                   const int* __restrict__ label,
                                                   const float* __restrict__ label_type,
                                                   const int* __restrict__ jlast,
                                                   const int* __restrict__ topk,
                                                   float* __restrict__ out,
                                                   float* __restrict__ accbuf,
                                                   int V, int S, int B) {
  __shared__ float red[256];
  __shared__ int lab[LBL];
  __shared__ float lt[LBL], lv[LBL];
  __shared__ int cnt[LBL];
  __shared__ float nrec_sh;
  int b = blockIdx.x, tid = threadIdx.x;
  const float* crow = Cos + (size_t)b * V;
  int jl = jlast[b];
  if (tid < LBL) {
    int la = label[((size_t)b * S + jl) * LBL + tid];
    lab[tid] = la;
    lt[tid] = label_type[((size_t)b * S + jl) * LBL + tid];
    lv[tid] = crow[la];
    cnt[tid] = 0;
  }
  __syncthreads();

  float s = 0.f;
  int local[LBL];
#pragma unroll
  for (int l = 0; l < LBL; ++l) local[l] = 0;
  for (int v = tid; v < V; v += 256) {
    float cv = crow[v];
    s += cv;
#pragma unroll
    for (int l = 0; l < LBL; ++l) {
      float x = lv[l];
      if (cv > x || (cv == x && v < lab[l])) local[l]++;
    }
  }
  red[tid] = s;
  __syncthreads();
  for (int st = 128; st > 0; st >>= 1) {
    if (tid < st) red[tid] += red[tid + st];
    __syncthreads();
  }
  if (tid == 0) out[B + b] = red[0] / (float)V;  // loss_neg
#pragma unroll
  for (int l = 0; l < LBL; ++l)
    if (local[l]) atomicAdd(&cnt[l], local[l]);
  if (tid == 0) {
    float lts = 0.f, lp = 0.f;
    for (int l = 0; l < LBL; ++l) {
      lts += lt[l];
      lp += lt[l] * (1.f - lv[l]);
    }
    out[b] = lp / (lts + EPSV);  // loss_pos
    nrec_sh = lts;
  }
  __syncthreads();
  if (tid == 0) {
    int K = topk[0];
    float hits = 0.f, nr = nrec_sh;
    for (int l = 0; l < LBL; ++l)
      if (cnt[l] < K && (float)l < nr) hits += 1.f;
    atomicAdd(&accbuf[0], hits);
    atomicAdd(&accbuf[1], nr);
  }
}

__global__ void finalize_kernel(const float* __restrict__ accbuf, float* __restrict__ out, int B) {
  out[2 * B]     = accbuf[0] / accbuf[1];
  out[2 * B + 1] = accbuf[1];
}

// ---------------------------------------------------------------------------
extern "C" void kernel_launch(void* const* d_in, const int* in_sizes, int n_in,
                              void* d_out, int out_size, void* d_ws, size_t ws_size,
                              hipStream_t stream) {
  (void)in_sizes; (void)n_in; (void)out_size; (void)ws_size;
  const int B = 32, S = 40, V = 50000, E = 256, H = 512, G = 4 * H, BS = B * S;

  const int*   inputs      = (const int*)d_in[0];
  const int*   label       = (const int*)d_in[1];
  const float* label_type  = (const float*)d_in[2];
  const int*   mask        = (const int*)d_in[3];
  const int*   topk        = (const int*)d_in[4];
  const float* graph_table = (const float*)d_in[5];
  const float* video_emb   = (const float*)d_in[6];
  const float* Wih0 = (const float*)d_in[7];
  const float* Whh0 = (const float*)d_in[8];
  const float* bih0 = (const float*)d_in[9];
  const float* bhh0 = (const float*)d_in[10];
  const float* Wih1 = (const float*)d_in[11];
  const float* Whh1 = (const float*)d_in[12];
  const float* bih1 = (const float*)d_in[13];
  const float* bhh1 = (const float*)d_in[14];
  const float* Wl   = (const float*)d_in[15];
  const float* bl   = (const float*)d_in[16];
  float* out = (float*)d_out;

  char* ws = (char*)d_ws;
  size_t off = 0;
  auto alloc = [&](size_t bytes) -> void* {
    void* p = ws + off;
    off += (bytes + 255) & ~(size_t)255;
    return p;
  };
  _Float16* Wih0h   = (_Float16*)alloc((size_t)G * E * 2);
  _Float16* Whh0h   = (_Float16*)alloc((size_t)G * H * 2);
  _Float16* Wih1h   = (_Float16*)alloc((size_t)G * H * 2);
  _Float16* Whh1h   = (_Float16*)alloc((size_t)G * H * 2);
  _Float16* x16     = (_Float16*)alloc((size_t)BS * E * 2);
  float*    xg0     = (float*)alloc((size_t)BS * G * 4);
  float*    xg1     = (float*)alloc((size_t)BS * G * 4);
  _Float16* h0seq16 = (_Float16*)alloc((size_t)BS * H * 2);
  float*    h1seq   = (float*)alloc((size_t)BS * H * 4);
  _Float16* hcur16  = (_Float16*)alloc((size_t)B * H * 2);
  float*    ccur    = (float*)alloc((size_t)B * H * 4);
  float*    zbuf    = (float*)alloc((size_t)B * G * 4);
  float*    olast   = (float*)alloc((size_t)B * E * 4);
  _Float16* olast16 = (_Float16*)alloc((size_t)B * E * 2);
  float*    onorm   = (float*)alloc((size_t)B * 4);
  float*    vnorm   = (float*)alloc((size_t)V * 4);
  _Float16* ve16    = (_Float16*)alloc((size_t)V * E * 2);
  float*    cosbuf  = (float*)alloc((size_t)B * V * 4);
  int*      jlast   = (int*)alloc((size_t)B * 4);
  float*    accbuf  = (float*)alloc(2 * 4);

  // weights -> f16 (once per launch; reused 40x by the recurrence)
  cvt_f32_to_f16<<<(G * E + 255) / 256, 256, 0, stream>>>(Wih0, Wih0h, G * E);
  cvt_f32_to_f16<<<(G * H + 255) / 256, 256, 0, stream>>>(Whh0, Whh0h, G * H);
  cvt_f32_to_f16<<<(G * H + 255) / 256, 256, 0, stream>>>(Wih1, Wih1h, G * H);
  cvt_f32_to_f16<<<(G * H + 255) / 256, 256, 0, stream>>>(Whh1, Whh1h, G * H);

  // embedding gather straight to f16
  embed_kernel<<<(BS * E + 255) / 256, 256, 0, stream>>>(inputs, mask, graph_table, x16, E, BS * E);

  // xg0 = x @ Wih0^T + bih0 + bhh0
  gemm_gates_wmma<<<dim3(G / 64, BS / 16), 32, 0, stream>>>(x16, E, Wih0h, bih0, bhh0, xg0, G);

  // layer 0 recurrence
  init_state<<<(B * H + 255) / 256, 256, 0, stream>>>(hcur16, ccur, B * H);
  for (int t = 0; t < S; ++t) {
    lstm_z_wmma<<<dim3(G / 64, B / 16), 32, 0, stream>>>(hcur16, Whh0h, H, xg0 + (size_t)t * G,
                                                         S * G, zbuf);
    lstm_gates<<<(B * H + 255) / 256, 256, 0, stream>>>(zbuf, hcur16, ccur, h0seq16,
                                                        (float*)nullptr, H, S, t, B * H);
  }

  // xg1 = h0seq @ Wih1^T + bih1 + bhh1
  gemm_gates_wmma<<<dim3(G / 64, BS / 16), 32, 0, stream>>>(h0seq16, H, Wih1h, bih1, bhh1, xg1, G);

  // layer 1 recurrence (keeps f32 sequence for the final projection)
  init_state<<<(B * H + 255) / 256, 256, 0, stream>>>(hcur16, ccur, B * H);
  for (int t = 0; t < S; ++t) {
    lstm_z_wmma<<<dim3(G / 64, B / 16), 32, 0, stream>>>(hcur16, Whh1h, H, xg1 + (size_t)t * G,
                                                         S * G, zbuf);
    lstm_gates<<<(B * H + 255) / 256, 256, 0, stream>>>(zbuf, hcur16, ccur, (_Float16*)nullptr,
                                                        h1seq, H, S, t, B * H);
  }

  // last valid step per batch, projection + norms
  jlast_kernel<<<1, 32, 0, stream>>>(mask, S, jlast);
  proj_kernel<<<(B * E + 255) / 256, 256, 0, stream>>>(h1seq, jlast, Wl, bl, olast, olast16,
                                                       H, E, S, B * E);
  rownorm_kernel<<<B, 256, 0, stream>>>(olast, E, onorm);
  rownorm_cvt_kernel<<<V, 256, 0, stream>>>(video_emb, E, vnorm, ve16);

  // cosine similarity against full table (WMMA f16, M=32 x N=50000 x K=256)
  cos_wmma<<<V / 16, 32, 0, stream>>>(olast16, ve16, onorm, vnorm, cosbuf, V, E);

  // losses + accuracy
  zero_f32<<<1, 256, 0, stream>>>(accbuf, 2);
  loss_kernel<<<B, 256, 0, stream>>>(cosbuf, label, label_type, jlast, topk, out, accbuf, V, S, B);
  finalize_kernel<<<1, 1, 0, stream>>>(accbuf, out, B);
}
